// SpatialCrossAttention_56229711839608
// MI455X (gfx1250) — matrix operand order repaired and checked by reading, not compiled, not run on hardware
//
#include <hip/hip_runtime.h>

typedef __attribute__((ext_vector_type(16))) __bf16          v16bf;
typedef __attribute__((ext_vector_type(16))) unsigned short  v16u;
typedef __attribute__((ext_vector_type(8)))  float           v8f;

#define BATCH 8
#define CCH   256
#define HW    4096
#define HEADS 8
#define HC    32   // channels per head

__device__ __forceinline__ unsigned short bfbits(float f) {
    unsigned u = __builtin_bit_cast(unsigned, f);
    return (unsigned short)((u + 0x7FFFu + ((u >> 16) & 1u)) >> 16);  // RNE
}
__device__ __forceinline__ unsigned pk2(float a, float b) {
    return (unsigned)bfbits(a) | ((unsigned)bfbits(b) << 16);
}
__device__ __forceinline__ __bf16 f2bf(float f) {
    unsigned short h = bfbits(f);
    return __builtin_bit_cast(__bf16, h);
}

// ---------------------------------------------------------------------------
// Y[n][o][l] = sum_c W[o][c] * X[n][c][l] + bias[o]
// X: (BATCH, CCH, HW) fp32, W: (CCH, CCH) fp32, Y: (BATCH, CCH, HW) fp32.
// Block: 256 threads (8 waves), tile 64(M) x 128(N), K stepped by 32.
// LDS tiles are stored FRAGMENT-MAJOR (ISA 7.12.2 per-lane layout) so each
// lane's fragment is a contiguous 32B run -> 2x ds_load_b128, no gathers.
//   A fragment position i (0..15): K = (i<8 ? i + 8*half : 16 + (i-8) + 8*half)
//   B fragment position i (0..15): K = (i<8 ? i + 8*half : 16 + (i-8) + 8*half)
// ---------------------------------------------------------------------------
__global__ void gemm_bias(const float* __restrict__ X, const float* __restrict__ W,
                          const float* __restrict__ bias, float* __restrict__ Y) {
    __shared__ alignas(32) unsigned short Af[4][32][16];  // 4 M-subtiles, per-lane frags
    __shared__ alignas(32) unsigned short Bf[8][32][16];  // 8 N-subtiles, per-lane frags

    const int n     = blockIdx.z;
    const int mBase = blockIdx.y * 64;
    const int nBase = blockIdx.x * 128;
    const int tid   = threadIdx.x;
    const int wave  = tid >> 5;
    const int lane  = tid & 31;
    const int half  = lane >> 4;
    const int l15   = lane & 15;
    const int wm    = (wave >> 2) * 32;     // 0 / 32
    const int wn    = (wave & 3) * 32;      // 0 / 32 / 64 / 96
    const int t0    = wm >> 4;              // A subtile pair base
    const int s0    = wn >> 4;              // B subtile pair base

    const float* Xn = X + (size_t)n * CCH * HW;

    v8f acc[2][2];
    acc[0][0] = (v8f){}; acc[0][1] = (v8f){};
    acc[1][0] = (v8f){}; acc[1][1] = (v8f){};

    // A staging task (constant across K loop except k0): one 8-float run each
    const int a_t    = tid >> 6;          // 0..3  M subtile
    const int a_lane = (tid >> 1) & 31;   // destination lane
    const int a_run  = tid & 1;           // 0: K=8h.., 1: K=16+8h..
    const int a_m    = mBase + 16 * a_t + (a_lane & 15);
    const int a_koff = 8 * (a_lane >> 4) + 16 * a_run;

    for (int k0 = 0; k0 < CCH; k0 += 32) {
        // ---- stage A: 256 threads x one 32B run -> one 16B LDS store
        {
            const float* wp = W + (size_t)a_m * CCH + k0 + a_koff;
            float4 w0 = *(const float4*)wp;
            float4 w1 = *(const float4*)(wp + 4);
            uint4 u;
            u.x = pk2(w0.x, w0.y); u.y = pk2(w0.z, w0.w);
            u.z = pk2(w1.x, w1.y); u.w = pk2(w1.z, w1.w);
            *(uint4*)&Af[a_t][a_lane][a_run * 8] = u;
        }
        // ---- stage B: 512 tasks (16 k-pairs x 32 col-quads), 2 per thread
        #pragma unroll
        for (int it = 0; it < 2; ++it) {
            const int task  = tid + (it << 8);
            const int kp    = task >> 5;          // 0..15 -> k = 2*kp
            const int nq    = task & 31;          // column quad
            const int k     = kp * 2;
            const int halfs = (kp >> 2) & 1;      // destination lane half
            const int idx   = (k & 7) + ((k >> 4) << 3); // fragment position (even)
            const float* xp = Xn + (size_t)(k0 + k) * HW + nBase + nq * 4;
            float4 r0 = *(const float4*)xp;
            float4 r1 = *(const float4*)(xp + HW);
            const float* p0 = (const float*)&r0;
            const float* p1 = (const float*)&r1;
            #pragma unroll
            for (int e = 0; e < 4; ++e) {
                const int nn     = nq * 4 + e;
                const int bt     = nn >> 4;
                const int lane_d = halfs * 16 + (nn & 15);
                *(unsigned*)&Bf[bt][lane_d][idx] = pk2(p0[e], p1[e]);
            }
            if (k0 + 32 < CCH) __builtin_prefetch(xp + (size_t)32 * HW, 0, 0);
        }
        __syncthreads();

        // ---- fragments: contiguous 32B per lane -> 2x ds_load_b128 each
        v16bf a0 = __builtin_bit_cast(v16bf, *(const v16u*)&Af[t0][lane][0]);
        v16bf a1 = __builtin_bit_cast(v16bf, *(const v16u*)&Af[t0 + 1][lane][0]);
        v16bf b0 = __builtin_bit_cast(v16bf, *(const v16u*)&Bf[s0][lane][0]);
        v16bf b1 = __builtin_bit_cast(v16bf, *(const v16u*)&Bf[s0 + 1][lane][0]);

        acc[0][0] = __builtin_amdgcn_wmma_f32_16x16x32_bf16(false, a0, false, b0,
                                                            (short)0, acc[0][0], false, false);
        acc[0][1] = __builtin_amdgcn_wmma_f32_16x16x32_bf16(false, a0, false, b1,
                                                            (short)0, acc[0][1], false, false);
        acc[1][0] = __builtin_amdgcn_wmma_f32_16x16x32_bf16(false, a1, false, b0,
                                                            (short)0, acc[1][0], false, false);
        acc[1][1] = __builtin_amdgcn_wmma_f32_16x16x32_bf16(false, a1, false, b1,
                                                            (short)0, acc[1][1], false, false);
        __syncthreads();
    }

    #pragma unroll
    for (int tm = 0; tm < 2; ++tm) {
        #pragma unroll
        for (int tn = 0; tn < 2; ++tn) {
            #pragma unroll
            for (int r = 0; r < 8; ++r) {
                const int row = mBase + wm + tm * 16 + r + half * 8;
                const int col = nBase + wn + tn * 16 + l15;
                Y[((size_t)n * CCH + row) * HW + col] = acc[tm][tn][r] + bias[row];
            }
        }
    }
}

// ---------------------------------------------------------------------------
// Softmax over the token axis (rows of 4096), in place.
// ---------------------------------------------------------------------------
__global__ void softmax_hw(float* __restrict__ X) {
    __shared__ float red[256];
    float* p = X + (size_t)blockIdx.x * HW;
    const int t = threadIdx.x;

    float m = -1e30f;
    for (int i = t; i < HW; i += 256) m = fmaxf(m, p[i]);
    red[t] = m; __syncthreads();
    for (int s = 128; s > 0; s >>= 1) {
        if (t < s) red[t] = fmaxf(red[t], red[t + s]);
        __syncthreads();
    }
    m = red[0]; __syncthreads();

    float sum = 0.f;
    for (int i = t; i < HW; i += 256) { float e = __expf(p[i] - m); p[i] = e; sum += e; }
    red[t] = sum; __syncthreads();
    for (int s = 128; s > 0; s >>= 1) {
        if (t < s) red[t] += red[t + s];
        __syncthreads();
    }
    const float inv = 1.0f / red[0];
    for (int i = t; i < HW; i += 256) p[i] *= inv;
}

// ---------------------------------------------------------------------------
// Softmax over the 32 per-head channels for each (n, head, token), in place.
// ---------------------------------------------------------------------------
__global__ void softmax_ch(float* __restrict__ X) {
    const size_t idx = (size_t)blockIdx.x * blockDim.x + threadIdx.x;
    const int l  = (int)(idx & (HW - 1));
    const int nh = (int)(idx >> 12);           // 0 .. 63
    float* p = X + ((size_t)nh * HC) * HW + l; // channel stride = HW

    float m = -1e30f;
    #pragma unroll
    for (int c = 0; c < HC; ++c) m = fmaxf(m, p[(size_t)c * HW]);
    float e[HC]; float s = 0.f;
    #pragma unroll
    for (int c = 0; c < HC; ++c) { e[c] = __expf(p[(size_t)c * HW] - m); s += e[c]; }
    const float inv = 1.0f / s;
    #pragma unroll
    for (int c = 0; c < HC; ++c) p[(size_t)c * HW] = e[c] * inv;
}

// ---------------------------------------------------------------------------
// context[nh][k][v] = sum_l keys[nh][k][l] * values[nh][v][l]
// grid = 64 (n*heads), block = 128 (4 waves: 2x2 tiles of 16x16), K loop 4096/32.
// Fragment builds are explicit float4 runs (A interleave = two 8-float runs).
// ---------------------------------------------------------------------------
__global__ void ctx_kernel(const float* __restrict__ Kp, const float* __restrict__ Vp,
                           float* __restrict__ C) {
    const int nh   = blockIdx.x;
    const int wave = threadIdx.x >> 5;
    const int lane = threadIdx.x & 31;
    const int half = lane >> 4;
    const int l15  = lane & 15;
    const int mOff = (wave >> 1) * 16;   // k-dim tile
    const int nOff = (wave & 1) * 16;    // v-dim tile

    const float* Kb = Kp + (size_t)nh * HC * HW;
    const float* Vb = Vp + (size_t)nh * HC * HW;

    v8f acc = (v8f){};
    for (int l0 = 0; l0 < HW; l0 += 32) {
        const float* arow = Kb + (size_t)(mOff + l15) * HW + l0 + half * 8;
        float4 fa0 = *(const float4*)arow;          // K = 8h .. 8h+3
        float4 fa1 = *(const float4*)(arow + 4);    // K = 8h+4 .. 8h+7
        float4 fa2 = *(const float4*)(arow + 16);   // K = 16+8h ..
        float4 fa3 = *(const float4*)(arow + 20);
        v16bf a;
        a[0]=f2bf(fa0.x); a[1]=f2bf(fa0.y); a[2]=f2bf(fa0.z); a[3]=f2bf(fa0.w);
        a[4]=f2bf(fa1.x); a[5]=f2bf(fa1.y); a[6]=f2bf(fa1.z); a[7]=f2bf(fa1.w);
        a[8]=f2bf(fa2.x); a[9]=f2bf(fa2.y); a[10]=f2bf(fa2.z); a[11]=f2bf(fa2.w);
        a[12]=f2bf(fa3.x); a[13]=f2bf(fa3.y); a[14]=f2bf(fa3.z); a[15]=f2bf(fa3.w);

        const float* brow = Vb + (size_t)(nOff + l15) * HW + l0 + half * 8;
        float4 fb0 = *(const float4*)brow;
        float4 fb1 = *(const float4*)(brow + 4);
        float4 fb2 = *(const float4*)(brow + 16);
        float4 fb3 = *(const float4*)(brow + 20);
        v16bf b;
        b[0]=f2bf(fb0.x); b[1]=f2bf(fb0.y); b[2]=f2bf(fb0.z); b[3]=f2bf(fb0.w);
        b[4]=f2bf(fb1.x); b[5]=f2bf(fb1.y); b[6]=f2bf(fb1.z); b[7]=f2bf(fb1.w);
        b[8]=f2bf(fb2.x); b[9]=f2bf(fb2.y); b[10]=f2bf(fb2.z); b[11]=f2bf(fb2.w);
        b[12]=f2bf(fb3.x); b[13]=f2bf(fb3.y); b[14]=f2bf(fb3.z); b[15]=f2bf(fb3.w);

        acc = __builtin_amdgcn_wmma_f32_16x16x32_bf16(false, a, false, b,
                                                      (short)0, acc, false, false);
    }

    float* Cb = C + (size_t)nh * HC * HC;
    #pragma unroll
    for (int r = 0; r < 8; ++r) {
        const int row = mOff + r + half * 8;   // k index
        const int col = nOff + l15;            // v index
        Cb[row * HC + col] = acc[r];
    }
}

// ---------------------------------------------------------------------------
// attended[nh][v][l] = sum_k ctx[nh][k][v] * Q[nh][k][l]   (single K=32 WMMA)
// grid = (64 col-groups, 64 nh), block = 128 (4 waves x 16 cols).
// ---------------------------------------------------------------------------
__global__ void att_kernel(const float* __restrict__ Ctx, const float* __restrict__ Q,
                           float* __restrict__ A) {
    const int nh   = blockIdx.y;
    const int wave = threadIdx.x >> 5;
    const int lane = threadIdx.x & 31;
    const int half = lane >> 4;
    const int l15  = lane & 15;
    const int col  = blockIdx.x * 64 + wave * 16 + l15;

    const float* Cb = Ctx + (size_t)nh * HC * HC;
    const float* Qb = Q + (size_t)nh * HC * HW;

    v16bf b;
    #pragma unroll
    for (int i = 0; i < 16; ++i)
        b[i] = f2bf(Qb[(size_t)((i & 7) + ((i >> 3) << 4) + half * 8) * HW + col]);

    #pragma unroll
    for (int t = 0; t < 2; ++t) {
        v16bf a;
        const int m = t * 16 + l15;            // v index (row of A = ctx^T)
        #pragma unroll
        for (int j = 0; j < 8; ++j) {
            int kb = ((j < 4) ? (2 * j) : (16 + 2 * (j - 4))) + half * 8;
            a[2 * j]     = f2bf(Cb[kb * HC + m]);
            a[2 * j + 1] = f2bf(Cb[(kb + 1) * HC + m]);
        }
        v8f acc = (v8f){};
        acc = __builtin_amdgcn_wmma_f32_16x16x32_bf16(false, a, false, b,
                                                      (short)0, acc, false, false);
        #pragma unroll
        for (int r = 0; r < 8; ++r) {
            const int v = t * 16 + r + half * 8;
            A[((size_t)nh * HC + v) * HW + col] = acc[r];
        }
    }
}

// ---------------------------------------------------------------------------
extern "C" void kernel_launch(void* const* d_in, const int* in_sizes, int n_in,
                              void* d_out, int out_size, void* d_ws, size_t ws_size,
                              hipStream_t stream) {
    (void)in_sizes; (void)n_in; (void)out_size; (void)ws_size;

    const float* input_  = (const float*)d_in[0];
    const float* input_T = (const float*)d_in[1];
    const float* Wk_rgb  = (const float*)d_in[2];  const float* bk_rgb = (const float*)d_in[3];
    const float* Wq_rgb  = (const float*)d_in[4];  const float* bq_rgb = (const float*)d_in[5];
    const float* Wv_rgb  = (const float*)d_in[6];  const float* bv_rgb = (const float*)d_in[7];
    const float* Wk_T    = (const float*)d_in[8];  const float* bk_T   = (const float*)d_in[9];
    const float* Wq_T    = (const float*)d_in[10]; const float* bq_T   = (const float*)d_in[11];
    const float* Wv_T    = (const float*)d_in[12]; const float* bv_T   = (const float*)d_in[13];
    const float* Wr_rgb  = (const float*)d_in[14]; const float* br_rgb = (const float*)d_in[15];
    const float* Wr_d    = (const float*)d_in[16]; const float* br_d   = (const float*)d_in[17];

    float* out = (float*)d_out;
    float* ws  = (float*)d_ws;

    const size_t PB = (size_t)BATCH * CCH * HW;  // 8,388,608 floats per buffer
    float* Pk_rgb = ws + 0 * PB;
    float* Pq_rgb = ws + 1 * PB;
    float* Pv_rgb = ws + 2 * PB;
    float* Pk_T   = ws + 3 * PB;
    float* Pq_T   = ws + 4 * PB;
    float* Pv_T   = ws + 5 * PB;
    float* Crgb   = ws + 6 * PB;
    float* CT     = Crgb + (size_t)BATCH * HEADS * HC * HC;

    dim3 gG(HW / 128, CCH / 64, BATCH);  // (32, 4, 8)
    dim3 bG(256);

    // 1) six projection GEMMs (fp32 in, bf16 WMMA, fp32 out, bias fused)
    gemm_bias<<<gG, bG, 0, stream>>>(input_,  Wk_rgb, bk_rgb, Pk_rgb);
    gemm_bias<<<gG, bG, 0, stream>>>(input_,  Wq_rgb, bq_rgb, Pq_rgb);
    gemm_bias<<<gG, bG, 0, stream>>>(input_,  Wv_rgb, bv_rgb, Pv_rgb);
    gemm_bias<<<gG, bG, 0, stream>>>(input_T, Wk_T,   bk_T,   Pk_T);
    gemm_bias<<<gG, bG, 0, stream>>>(input_T, Wq_T,   bq_T,   Pq_T);
    gemm_bias<<<gG, bG, 0, stream>>>(input_T, Wv_T,   bv_T,   Pv_T);

    // 2) softmaxes (keys over tokens, queries over head channels), in place
    softmax_hw<<<BATCH * CCH, 256, 0, stream>>>(Pk_rgb);
    softmax_hw<<<BATCH * CCH, 256, 0, stream>>>(Pk_T);
    softmax_ch<<<(BATCH * HEADS * HW) / 256, 256, 0, stream>>>(Pq_rgb);
    softmax_ch<<<(BATCH * HEADS * HW) / 256, 256, 0, stream>>>(Pq_T);

    // 3) per-head contexts (32x32) via WMMA over 4096 tokens
    ctx_kernel<<<BATCH * HEADS, 128, 0, stream>>>(Pk_rgb, Pv_rgb, Crgb);
    ctx_kernel<<<BATCH * HEADS, 128, 0, stream>>>(Pk_T,   Pv_T,   CT);

    // 4) cross-attended maps; spent key buffers are reused as outputs
    att_kernel<<<dim3(HW / 64, BATCH * HEADS), 128, 0, stream>>>(CT,   Pq_rgb, Pk_rgb);
    att_kernel<<<dim3(HW / 64, BATCH * HEADS), 128, 0, stream>>>(Crgb, Pq_T,   Pk_T);

    // 5) re-attention 1x1 convs straight into d_out (rgb first, then thermal)
    gemm_bias<<<gG, bG, 0, stream>>>(Pk_rgb, Wr_rgb, br_rgb, out);
    gemm_bias<<<gG, bG, 0, stream>>>(Pk_T,   Wr_d,   br_d,   out + PB);
}
